// SequentialDynamicEdgeConv_47321949667505
// MI455X (gfx1250) — compile-verified, matrix operand order checked by reading, not run on hardware
//
#include <hip/hip_runtime.h>

#define BB 16
#define PP 2048
#define KNN 20
#define NN (BB * PP)          // 32768 points
#define EE (NN * KNN)         // 655360 edges
#define EPSV 1e-5f

typedef __attribute__((ext_vector_type(16))) _Float16 v16h;
typedef __attribute__((ext_vector_type(8)))  float    v8f;

// K index inside a 16x32 (MxK) f16 WMMA A/B fragment for (lane, half h)
__device__ __forceinline__ int wmma_kidx(int lane, int h) {
  return ((h >> 3) << 4) + (h & 7) + ((lane & 16) ? 8 : 0);
}

// ---------------------------------------------------------------------------
// Weight pre-swizzle: W[Kdim][Ndim] (row-major, in->out) -> B-fragment layout.
// frag element t = ((nt*Nkt + kt)*32 + lane)*16 + h holds W[k][n],
//   n = nt*16 + lane%16,  k = kt*32 + wmma_kidx(lane,h)  (zero-padded K).
// ---------------------------------------------------------------------------
__global__ void pack_w_k(const float* __restrict__ W, _Float16* __restrict__ out,
                         int Kdim, int Nkt, int Ndim) {
  int t = blockIdx.x * 256 + threadIdx.x;
  int total = (Ndim / 16) * Nkt * 512;
  if (t >= total) return;
  int h = t & 15;
  int lane = (t >> 4) & 31;
  int tl = t >> 9;
  int kt = tl % Nkt;
  int nt = tl / Nkt;
  int n = nt * 16 + (lane & 15);
  int k = kt * 32 + wmma_kidx(lane, h);
  float v = (k < Kdim) ? W[k * Ndim + n] : 0.0f;
  out[t] = (_Float16)v;
}

// ---------------------------------------------------------------------------
// Input BN: stats (sum / sumsq per 4 channels) then apply.
// stats block layout (per BN instance, 512 floats):
//   [0..127]=sum, [128..255]=sumsq, [256..383]=scale, [384..511]=shift
// ---------------------------------------------------------------------------
__global__ void bnin_stats_k(const float* __restrict__ x, float* __restrict__ st) {
  __shared__ float ls[8];
  if (threadIdx.x < 8) ls[threadIdx.x] = 0.0f;
  __syncthreads();
  float s0 = 0, s1 = 0, s2 = 0, s3 = 0, q0 = 0, q1 = 0, q2 = 0, q3 = 0;
  for (int i = blockIdx.x * blockDim.x + threadIdx.x; i < NN;
       i += gridDim.x * blockDim.x) {
    float v0 = x[i * 4 + 0], v1 = x[i * 4 + 1];
    float v2 = x[i * 4 + 2], v3 = x[i * 4 + 3];
    s0 += v0; q0 += v0 * v0;
    s1 += v1; q1 += v1 * v1;
    s2 += v2; q2 += v2 * v2;
    s3 += v3; q3 += v3 * v3;
  }
  atomicAdd(&ls[0], s0); atomicAdd(&ls[1], s1);
  atomicAdd(&ls[2], s2); atomicAdd(&ls[3], s3);
  atomicAdd(&ls[4], q0); atomicAdd(&ls[5], q1);
  atomicAdd(&ls[6], q2); atomicAdd(&ls[7], q3);
  __syncthreads();
  if (threadIdx.x < 4) {
    atomicAdd(&st[threadIdx.x], ls[threadIdx.x]);
    atomicAdd(&st[128 + threadIdx.x], ls[4 + threadIdx.x]);
  }
}

__global__ void bnin_apply_k(const float* __restrict__ x,
                             const float* __restrict__ st,
                             float* __restrict__ xn) {
  int i = blockIdx.x * 256 + threadIdx.x;
  if (i >= NN * 4) return;
  int c = i & 3;
  xn[i] = x[i] * st[256 + c] + st[384 + c];
}

// Fold (sum,sumsq,count,gamma,beta) -> (scale,shift) for y*scale+shift
__global__ void finalize_k(float* __restrict__ st, const float* __restrict__ g,
                           const float* __restrict__ beta, int nch, float invn) {
  int c = threadIdx.x;
  if (c >= nch) return;
  float mean = st[c] * invn;
  float var = st[128 + c] * invn - mean * mean;
  float sc = g[c] * rsqrtf(var + EPSV);
  st[256 + c] = sc;
  st[384 + c] = beta[c] - mean * sc;
}

// ---------------------------------------------------------------------------
// kNN: per-thread point, LDS-tiled candidates, register-resident top-20.
// ---------------------------------------------------------------------------
template <int C, int TQ>
__global__ void __launch_bounds__(128) knn_k(const float* __restrict__ xin,
                                             int* __restrict__ idx) {
  __shared__ float qbuf[TQ * C];
  const int b = blockIdx.y;
  const int p = blockIdx.x * 128 + threadIdx.x;
  const float* xb = xin + (long)b * PP * C;
  float xp[C];
#pragma unroll
  for (int c = 0; c < C; ++c) xp[c] = xb[(long)p * C + c];
  float dk[KNN];
  int ik[KNN];
#pragma unroll
  for (int k = 0; k < KNN; ++k) { dk[k] = 3.4e38f; ik[k] = 0; }
  for (int q0 = 0; q0 < PP; q0 += TQ) {
    __syncthreads();
    for (int i = threadIdx.x; i < TQ * C; i += 128) qbuf[i] = xb[(long)q0 * C + i];
    __syncthreads();
    for (int qq = 0; qq < TQ; ++qq) {
      float d = 0.0f;
#pragma unroll
      for (int c = 0; c < C; ++c) {
        float t = xp[c] - qbuf[qq * C + c];
        d = fmaf(t, t, d);
      }
      if (d < dk[KNN - 1]) {
        float cd = d;
        int ci = q0 + qq;
#pragma unroll
        for (int k = 0; k < KNN; ++k) {
          if (cd < dk[k]) {
            float td = dk[k]; int ti = ik[k];
            dk[k] = cd; ik[k] = ci;
            cd = td; ci = ti;
          }
        }
      }
    }
  }
  int base = ((long)b * PP + p) * KNN;
#pragma unroll
  for (int k = 0; k < KNN; ++k) idx[base + k] = ik[k];
}

// ---------------------------------------------------------------------------
// EdgeConv MLP via WMMA. 4 waves/block, each wave owns a 16-edge tile.
//   MODE 0: y1 = e@w1+b1, accumulate bn1 stats
//   MODE 1: h1 = relu(bn1(y1)); y2 = h1@w2+b2, accumulate bn2 stats
//   MODE 2: h2 = relu(bn2(y2)); atomicMax over K into xout (N x H2)
// ---------------------------------------------------------------------------
template <int CIN, int KP1, int H1, int H2, int MODE>
__global__ void __launch_bounds__(128) edgeconv_k(
    const float* __restrict__ xin, const int* __restrict__ idx,
    const _Float16* __restrict__ w1f, const float* __restrict__ b1,
    const _Float16* __restrict__ w2f, const float* __restrict__ b2,
    float* st1, float* st2, float* __restrict__ xout) {
  __shared__ _Float16 e_lds[4][16][KP1];
  __shared__ _Float16 h_lds[4][16][H1];
  __shared__ float sstat[256];
  const int lane = threadIdx.x & 31;
  const int wid = threadIdx.x >> 5;
  if (MODE != 2) {
    for (int t = threadIdx.x; t < 256; t += 128) sstat[t] = 0.0f;
  }
  const int e0 = (blockIdx.x * 4 + wid) * 16;
  const int r = lane & 15;
  const int edge = e0 + r;
  const int pg = edge / KNN;  // b*P + p
  const int q = idx[edge];
  const int qg = (pg / PP) * PP + q;
  const float* xi = xin + (long)pg * CIN;
  const float* xj = xin + (long)qg * CIN;
  if (lane < 16) {
#pragma unroll
    for (int c = 0; c < CIN; ++c) e_lds[wid][r][c] = (_Float16)xi[c];
#pragma unroll
    for (int c = 2 * CIN; c < KP1; ++c) e_lds[wid][r][c] = (_Float16)0.0f;
  } else {
#pragma unroll
    for (int c = 0; c < CIN; ++c)
      e_lds[wid][r][CIN + c] = (_Float16)(xj[c] - xi[c]);
  }
  __syncthreads();

  // ---- GEMM 1: (16 x KP1) @ (KP1 x H1) ----
  constexpr int NKT1 = KP1 / 32, NT1 = H1 / 16;
#pragma unroll
  for (int nt = 0; nt < NT1; ++nt) {
    v8f acc = {};
#pragma unroll
    for (int kt = 0; kt < NKT1; ++kt) {
      v16h a;
#pragma unroll
      for (int h = 0; h < 16; ++h)
        a[h] = e_lds[wid][r][kt * 32 + wmma_kidx(lane, h)];
      v16h bf = *(const v16h*)(w1f + ((nt * NKT1 + kt) * 32 + lane) * 16);
      acc = __builtin_amdgcn_wmma_f32_16x16x32_f16(false, a, false, bf,
                                                   (short)0, acc, false, false);
    }
    const int ch = nt * 16 + r;
    const float bias = b1[ch];
    if (MODE == 0) {
      float s = 0.0f, ss = 0.0f;
#pragma unroll
      for (int j = 0; j < 8; ++j) {
        float v = acc[j] + bias;
        s += v; ss += v * v;
      }
      s += __shfl_xor(s, 16, 32);
      ss += __shfl_xor(ss, 16, 32);
      if (lane < 16) { atomicAdd(&sstat[ch], s); atomicAdd(&sstat[128 + ch], ss); }
    } else {
      const float sc = st1[256 + ch], sh = st1[384 + ch];
#pragma unroll
      for (int j = 0; j < 8; ++j) {
        float v = fmaxf((acc[j] + bias) * sc + sh, 0.0f);
        int row = j + ((lane & 16) ? 8 : 0);
        h_lds[wid][row][ch] = (_Float16)v;
      }
    }
  }

  // ---- GEMM 2: (16 x H1) @ (H1 x H2) ----
  if (MODE >= 1) {
    __syncthreads();
    constexpr int NKT2 = H1 / 32, NT2 = H2 / 16;
#pragma unroll
    for (int nt = 0; nt < NT2; ++nt) {
      v8f acc = {};
#pragma unroll
      for (int kt = 0; kt < NKT2; ++kt) {
        v16h a;
#pragma unroll
        for (int h = 0; h < 16; ++h)
          a[h] = h_lds[wid][r][kt * 32 + wmma_kidx(lane, h)];
        v16h bf = *(const v16h*)(w2f + ((nt * NKT2 + kt) * 32 + lane) * 16);
        acc = __builtin_amdgcn_wmma_f32_16x16x32_f16(false, a, false, bf,
                                                     (short)0, acc, false, false);
      }
      const int ch = nt * 16 + r;
      const float bias = b2[ch];
      if (MODE == 1) {
        float s = 0.0f, ss = 0.0f;
#pragma unroll
        for (int j = 0; j < 8; ++j) {
          float v = acc[j] + bias;
          s += v; ss += v * v;
        }
        s += __shfl_xor(s, 16, 32);
        ss += __shfl_xor(ss, 16, 32);
        if (lane < 16) { atomicAdd(&sstat[ch], s); atomicAdd(&sstat[128 + ch], ss); }
      } else {
        const float sc = st2[256 + ch], sh = st2[384 + ch];
#pragma unroll
        for (int j = 0; j < 8; ++j) {
          float v = fmaxf((acc[j] + bias) * sc + sh, 0.0f);
          int row = j + ((lane & 16) ? 8 : 0);
          int pgo = (e0 + row) / KNN;  // point of this edge row
          atomicMax((int*)(xout + (long)pgo * H2 + ch), __float_as_int(v));
        }
      }
    }
  }

  if (MODE != 2) {
    __syncthreads();
    float* dst = (MODE == 0) ? st1 : st2;
    constexpr int HX = (MODE == 0) ? H1 : H2;
    for (int t = threadIdx.x; t < HX; t += 128) {
      atomicAdd(&dst[t], sstat[t]);
      atomicAdd(&dst[128 + t], sstat[128 + t]);
    }
  }
}

// ---------------------------------------------------------------------------
// Classifier: xc = [x1|x2] (N x 192) @ cl_w1 (192 x 128), BN+ReLU, @ cl_w2.
// ---------------------------------------------------------------------------
template <int MODE>
__global__ void __launch_bounds__(128) classifier_k(
    const float* __restrict__ x1, const float* __restrict__ x2,
    const _Float16* __restrict__ w1f, const float* __restrict__ b1, float* st,
    const float* __restrict__ w2, const float* __restrict__ b2,
    float* __restrict__ out) {
  constexpr int KP = 192, H = 128, NKT = KP / 32, NT = H / 16;
  __shared__ _Float16 xc[4][16][KP];
  __shared__ _Float16 h_lds[4][16][H];
  __shared__ float sstat[256];
  const int lane = threadIdx.x & 31;
  const int wid = threadIdx.x >> 5;
  if (MODE == 0) {
    for (int t = threadIdx.x; t < 256; t += 128) sstat[t] = 0.0f;
  }
  const int row0 = (blockIdx.x * 4 + wid) * 16;
  const int r = lane & 15;
  const int rowg = row0 + r;
  if (lane < 16) {
#pragma unroll
    for (int c = 0; c < 64; ++c) xc[wid][r][c] = (_Float16)x1[(long)rowg * 64 + c];
  } else {
#pragma unroll
    for (int c = 0; c < 128; ++c)
      xc[wid][r][64 + c] = (_Float16)x2[(long)rowg * 128 + c];
  }
  __syncthreads();

#pragma unroll
  for (int nt = 0; nt < NT; ++nt) {
    v8f acc = {};
#pragma unroll
    for (int kt = 0; kt < NKT; ++kt) {
      v16h a;
#pragma unroll
      for (int h = 0; h < 16; ++h)
        a[h] = xc[wid][r][kt * 32 + wmma_kidx(lane, h)];
      v16h bf = *(const v16h*)(w1f + ((nt * NKT + kt) * 32 + lane) * 16);
      acc = __builtin_amdgcn_wmma_f32_16x16x32_f16(false, a, false, bf,
                                                   (short)0, acc, false, false);
    }
    const int ch = nt * 16 + r;
    const float bias = b1[ch];
    if (MODE == 0) {
      float s = 0.0f, ss = 0.0f;
#pragma unroll
      for (int j = 0; j < 8; ++j) {
        float v = acc[j] + bias;
        s += v; ss += v * v;
      }
      s += __shfl_xor(s, 16, 32);
      ss += __shfl_xor(ss, 16, 32);
      if (lane < 16) { atomicAdd(&sstat[ch], s); atomicAdd(&sstat[128 + ch], ss); }
    } else {
      const float sc = st[256 + ch], sh = st[384 + ch];
#pragma unroll
      for (int j = 0; j < 8; ++j) {
        float v = fmaxf((acc[j] + bias) * sc + sh, 0.0f);
        int row = j + ((lane & 16) ? 8 : 0);
        h_lds[wid][row][ch] = (_Float16)v;
      }
    }
  }

  if (MODE == 0) {
    __syncthreads();
    for (int t = threadIdx.x; t < H; t += 128) {
      atomicAdd(&st[t], sstat[t]);
      atomicAdd(&st[128 + t], sstat[128 + t]);
    }
  } else {
    __syncthreads();
    const int half = lane >> 4;  // split the 128-dot across lane pairs
    float a2 = 0.0f;
#pragma unroll
    for (int c = 0; c < 64; ++c)
      a2 += (float)h_lds[wid][r][half * 64 + c] * w2[half * 64 + c];
    a2 += __shfl_xor(a2, 16, 32);
    if (lane < 16) out[rowg] = a2 + b2[0];
  }
}

// ---------------------------------------------------------------------------
extern "C" void kernel_launch(void* const* d_in, const int* in_sizes, int n_in,
                              void* d_out, int out_size, void* d_ws,
                              size_t ws_size, hipStream_t stream) {
  (void)in_sizes; (void)n_in; (void)out_size; (void)ws_size;
  const float* x      = (const float*)d_in[0];
  const float* in_g   = (const float*)d_in[2];
  const float* in_b   = (const float*)d_in[3];
  const float* c1_w1  = (const float*)d_in[4];
  const float* c1_b1  = (const float*)d_in[5];
  const float* c1_g1  = (const float*)d_in[6];
  const float* c1_bb1 = (const float*)d_in[7];
  const float* c1_w2  = (const float*)d_in[8];
  const float* c1_b2  = (const float*)d_in[9];
  const float* c1_g2  = (const float*)d_in[10];
  const float* c1_bb2 = (const float*)d_in[11];
  const float* c2_w1  = (const float*)d_in[12];
  const float* c2_b1  = (const float*)d_in[13];
  const float* c2_g1  = (const float*)d_in[14];
  const float* c2_bb1 = (const float*)d_in[15];
  const float* c2_w2  = (const float*)d_in[16];
  const float* c2_b2  = (const float*)d_in[17];
  const float* c2_g2  = (const float*)d_in[18];
  const float* c2_bb2 = (const float*)d_in[19];
  const float* cl_w1  = (const float*)d_in[20];
  const float* cl_b1  = (const float*)d_in[21];
  const float* cl_g   = (const float*)d_in[22];
  const float* cl_bb  = (const float*)d_in[23];
  const float* cl_w2  = (const float*)d_in[24];
  const float* cl_b2  = (const float*)d_in[25];
  float* out = (float*)d_out;

  char* ws = (char*)d_ws;
  size_t off = 0;
  auto alloc = [&](size_t bytes) -> void* {
    void* p = (void*)(ws + off);
    off += (bytes + 255) & ~(size_t)255;
    return p;
  };
  float* xn      = (float*)alloc((size_t)NN * 4 * 4);
  float* x1      = (float*)alloc((size_t)NN * 64 * 4);
  float* x2      = (float*)alloc((size_t)NN * 128 * 4);
  int*   idx1    = (int*)alloc((size_t)EE * 4);
  int*   idx2    = (int*)alloc((size_t)EE * 4);
  float* st      = (float*)alloc((size_t)6 * 512 * 4);
  _Float16* c1w1f = (_Float16*)alloc((size_t)(64 / 16) * 1 * 512 * 2);
  _Float16* c1w2f = (_Float16*)alloc((size_t)(64 / 16) * 2 * 512 * 2);
  _Float16* c2w1f = (_Float16*)alloc((size_t)(128 / 16) * 4 * 512 * 2);
  _Float16* c2w2f = (_Float16*)alloc((size_t)(128 / 16) * 4 * 512 * 2);
  _Float16* clw1f = (_Float16*)alloc((size_t)(128 / 16) * 6 * 512 * 2);

  float* st0 = st + 0 * 512;  // input BN (4 ch)
  float* st1 = st + 1 * 512;  // conv1 bn1 (64)
  float* st2 = st + 2 * 512;  // conv1 bn2 (64)
  float* st3 = st + 3 * 512;  // conv2 bn1 (128)
  float* st4 = st + 4 * 512;  // conv2 bn2 (128)
  float* st5 = st + 5 * 512;  // classifier bn (128)

  // Zero max-accumulators (ReLU outputs >= 0 => int atomicMax trick) + stats.
  hipMemsetAsync(x1, 0, (size_t)NN * 64 * 4, stream);
  hipMemsetAsync(x2, 0, (size_t)NN * 128 * 4, stream);
  hipMemsetAsync(st, 0, (size_t)6 * 512 * 4, stream);

  // Pre-swizzle weights into WMMA B-fragment layout (f16).
  pack_w_k<<<(4 * 1 * 512 + 255) / 256, 256, 0, stream>>>(c1_w1, c1w1f, 8, 1, 64);
  pack_w_k<<<(4 * 2 * 512 + 255) / 256, 256, 0, stream>>>(c1_w2, c1w2f, 64, 2, 64);
  pack_w_k<<<(8 * 4 * 512 + 255) / 256, 256, 0, stream>>>(c2_w1, c2w1f, 128, 4, 128);
  pack_w_k<<<(8 * 4 * 512 + 255) / 256, 256, 0, stream>>>(c2_w2, c2w2f, 128, 4, 128);
  pack_w_k<<<(8 * 6 * 512 + 255) / 256, 256, 0, stream>>>(cl_w1, clw1f, 192, 6, 128);

  // Input BN.
  bnin_stats_k<<<256, 256, 0, stream>>>(x, st0);
  finalize_k<<<1, 128, 0, stream>>>(st0, in_g, in_b, 4, 1.0f / NN);
  bnin_apply_k<<<(NN * 4 + 255) / 256, 256, 0, stream>>>(x, st0, xn);

  // EdgeConv 1 (C=4).
  knn_k<4, 256><<<dim3(PP / 128, BB), 128, 0, stream>>>(xn, idx1);
  edgeconv_k<4, 32, 64, 64, 0><<<EE / 64, 128, 0, stream>>>(
      xn, idx1, c1w1f, c1_b1, c1w2f, c1_b2, st1, st2, x1);
  finalize_k<<<1, 128, 0, stream>>>(st1, c1_g1, c1_bb1, 64, 1.0f / EE);
  edgeconv_k<4, 32, 64, 64, 1><<<EE / 64, 128, 0, stream>>>(
      xn, idx1, c1w1f, c1_b1, c1w2f, c1_b2, st1, st2, x1);
  finalize_k<<<1, 128, 0, stream>>>(st2, c1_g2, c1_bb2, 64, 1.0f / EE);
  edgeconv_k<4, 32, 64, 64, 2><<<EE / 64, 128, 0, stream>>>(
      xn, idx1, c1w1f, c1_b1, c1w2f, c1_b2, st1, st2, x1);

  // EdgeConv 2 (C=64).
  knn_k<64, 64><<<dim3(PP / 128, BB), 128, 0, stream>>>(x1, idx2);
  edgeconv_k<64, 128, 128, 128, 0><<<EE / 64, 128, 0, stream>>>(
      x1, idx2, c2w1f, c2_b1, c2w2f, c2_b2, st3, st4, x2);
  finalize_k<<<1, 128, 0, stream>>>(st3, c2_g1, c2_bb1, 128, 1.0f / EE);
  edgeconv_k<64, 128, 128, 128, 1><<<EE / 64, 128, 0, stream>>>(
      x1, idx2, c2w1f, c2_b1, c2w2f, c2_b2, st3, st4, x2);
  finalize_k<<<1, 128, 0, stream>>>(st4, c2_g2, c2_bb2, 128, 1.0f / EE);
  edgeconv_k<64, 128, 128, 128, 2><<<EE / 64, 128, 0, stream>>>(
      x1, idx2, c2w1f, c2_b1, c2w2f, c2_b2, st3, st4, x2);

  // Classifier head.
  classifier_k<0><<<NN / 64, 128, 0, stream>>>(x1, x2, clw1f, cl_b1, st5,
                                               cl_w2, cl_b2, out);
  finalize_k<<<1, 128, 0, stream>>>(st5, cl_g, cl_bb, 128, 1.0f / NN);
  classifier_k<1><<<NN / 64, 128, 0, stream>>>(x1, x2, clw1f, cl_b1, st5,
                                               cl_w2, cl_b2, out);
}
